// LSTMEncoder_13469017440701
// MI455X (gfx1250) — matrix operand order repaired and checked by reading, not compiled, not run on hardware
//
#include <hip/hip_runtime.h>
#include <stdint.h>

// ---------------------------------------------------------------------------
// Problem constants (from reference)
// ---------------------------------------------------------------------------
#define BATCH   64
#define SRCLEN  1024
#define EMBED   512
#define HIDDEN  1024
#define GATES   (4 * HIDDEN)          // 4096
#define BT      (BATCH * SRCLEN)      // 65536

typedef unsigned short u16;
typedef __attribute__((ext_vector_type(16))) unsigned short v16u;
typedef __attribute__((ext_vector_type(16))) __bf16         v16bf;
typedef __attribute__((ext_vector_type(8)))  float          v8f;
typedef __attribute__((ext_vector_type(4)))  unsigned int   u32x4;
typedef __attribute__((ext_vector_type(8)))  int            i32x8;
typedef __attribute__((ext_vector_type(4)))  int            i32x4;

#if __has_builtin(__builtin_amdgcn_tensor_load_to_lds)
#define HAVE_TDM 1
#else
#define HAVE_TDM 0
#endif

// ---------------------------------------------------------------------------
// Helpers
// ---------------------------------------------------------------------------
__device__ __forceinline__ u16 f2bf(float f) {
    uint32_t x = __builtin_bit_cast(uint32_t, f);
    uint32_t r = x + 0x7FFFu + ((x >> 16) & 1u);   // round-to-nearest-even
    return (u16)(r >> 16);
}
__device__ __forceinline__ float bf2f(u16 u) {
    return __builtin_bit_cast(float, (uint32_t)u << 16);
}
__device__ __forceinline__ v8f vzero8() {
    v8f z = {0.f, 0.f, 0.f, 0.f, 0.f, 0.f, 0.f, 0.f};
    return z;
}
__device__ __forceinline__ v8f wmma_bf16(v16u a, v16u b, v8f c) {
    return __builtin_amdgcn_wmma_f32_16x16x32_bf16(
        false, __builtin_bit_cast(v16bf, a),
        false, __builtin_bit_cast(v16bf, b),
        (short)0, c, false, false);
}
__device__ __forceinline__ float sigmoidf(float x) {
    return 1.0f / (1.0f + __expf(-x));
}

// A-fragment (16x32 bf16, M x K) packing coordinates per CDNA5 ISA layout.
__device__ __forceinline__ void a_slot(int m, int kk, int* lane, int* elem) {
    int half  = ((kk & 15) >= 8) ? 16 : 0;
    int v     = ((kk >= 16) ? 4 : 0) + ((kk & 7) >> 1);
    *lane     = half + (m & 15);
    *elem     = v * 2 + (kk & 1);
}

#if HAVE_TDM
// ---------------------------------------------------------------------------
// Tensor Data Mover: 2D tile load Global -> LDS.
//   tile: `rows` rows of `row_bytes` bytes, global row stride `stride_bytes`,
//   packed contiguously into LDS at lds_dst. data_size = 8 bytes.
// D# layout per CDNA5 ISA (group0 128b, group1 256b).
// ---------------------------------------------------------------------------
__device__ __forceinline__ void tdm_load_2d(void* lds_dst, const void* gsrc,
                                            uint32_t row_bytes, uint32_t rows,
                                            uint32_t stride_bytes) {
    uint64_t ga = (uint64_t)(uintptr_t)gsrc;
    u32x4 g0;
    g0.x = 1u;                                               // count=1 (valid D#)
    g0.y = (uint32_t)(uintptr_t)lds_dst;                     // lds_addr
    g0.z = (uint32_t)(ga & 0xFFFFFFFFu);                     // global_addr[31:0]
    g0.w = (uint32_t)((ga >> 32) & 0x1FFFFFFu) | (2u << 30); // addr[56:32]|type=2

    uint32_t d0u  = row_bytes >> 3;                          // dim0 units of 8B
    uint32_t strd = stride_bytes >> 3;                       // stride units of 8B
    uint32_t td0  = 0x100000u;                               // large tensor_dim0
    i32x8 g1 = {0, 0, 0, 0, 0, 0, 0, 0};
    g1[0] = (int)(3u << 16);                                 // data_size = 8B
    g1[1] = (int)((td0 & 0xFFFFu) << 16);                    // tensor_dim0 lo16
    g1[2] = (int)((td0 >> 16) | (rows << 16));               // dim0 hi | tensor_dim1 lo
    g1[3] = (int)(d0u << 16);                                // tile_dim0
    g1[4] = (int)rows;                                       // tile_dim1
    g1[5] = (int)strd;                                       // tensor_dim0_stride lo32
    i32x4 z4 = {0, 0, 0, 0};
#if __clang_major__ >= 23
    i32x8 z8 = {0, 0, 0, 0, 0, 0, 0, 0};
    __builtin_amdgcn_tensor_load_to_lds(g0, g1, z4, z4, z8, 0);
#else
    __builtin_amdgcn_tensor_load_to_lds(g0, g1, z4, z4, 0);
#endif
}
#endif

// ---------------------------------------------------------------------------
// Pack a (N x K) row-major f32 weight into bf16 B-fragments (1 KB / 16x32 tile).
// ---------------------------------------------------------------------------
__global__ void pack_w_kernel(const float* __restrict__ W, u16* __restrict__ out,
                              int ntiles, int ktiles, int K) {
    int gid   = blockIdx.x * blockDim.x + threadIdx.x;
    int total = ntiles * ktiles * 32;
    if (gid >= total) return;
    int frag  = gid >> 5;
    int lane  = gid & 31;
    int ntile = frag / ktiles;
    int ktile = frag % ktiles;
    int ng    = ntile * 16 + (lane & 15);
    u16* dst  = out + (size_t)frag * 512 + lane * 16;
#pragma unroll
    for (int i = 0; i < 16; ++i) {
        int kg = ktile * 32 + ((i < 8) ? (8 * (lane >> 4) + i) : (16 + 8 * (lane >> 4) + (i - 8)));
        dst[i] = f2bf(W[(size_t)ng * K + kg]);
    }
}

// ---------------------------------------------------------------------------
// Zero the double-buffered packed-h state and d_out (h, c).
// ---------------------------------------------------------------------------
__global__ void init_state_kernel(u16* __restrict__ hA, float* __restrict__ out) {
    int i = blockIdx.x * blockDim.x + threadIdx.x;
    if (i < 2 * BATCH * HIDDEN) hA[i] = 0;
    if (i < 2 * BATCH * HIDDEN) out[i & (2 * BATCH * HIDDEN - 1)] = 0.0f;
}

// ---------------------------------------------------------------------------
// x_proj: (BT x EMBED) gathered-embedding GEMM vs packed Wx -> bf16 x_proj.
// Grid: (BT/64, GATES/128), block 256 (8 waves).
// A (embedding rows) staged to LDS by all threads (f32 -> bf16 convert);
// B (8 Wx fragments, 1 KB each @ 16 KB stride) DMA'd by the TDM in parallel.
// ---------------------------------------------------------------------------
__global__ void __launch_bounds__(256)
xproj_kernel(const int* __restrict__ src, const float* __restrict__ emb,
             const u16* __restrict__ WxP, const float* __restrict__ bx,
             u16* __restrict__ xp) {
    __shared__ __align__(32) u16 aSh[4][32][16];   // 4 M-tile A fragments (4 KB)
#if HAVE_TDM
    __shared__ __align__(32) u16 bSh[8][512];      // 8 B fragments (8 KB)
#endif

    const int tid    = threadIdx.x;
    const int lane   = tid & 31;
    const int wave   = tid >> 5;
    const int mblock = blockIdx.x * 64;
    const int ntile  = blockIdx.y * 8 + wave;

    v8f acc[4];
#pragma unroll
    for (int mt = 0; mt < 4; ++mt) acc[mt] = vzero8();

    for (int kt = 0; kt < EMBED / 32; ++kt) {
#if HAVE_TDM
        // Kick off the B-tile DMA (8 rows x 1 KB, 16 KB stride) from wave 0.
        if (wave == 0) {
            const u16* gsrc = WxP + ((size_t)(blockIdx.y * 8 * (EMBED / 32) + kt)) * 512;
            tdm_load_2d(&bSh[0][0], gsrc, 1024u, 8u, (uint32_t)(EMBED / 32) * 1024u);
        }
#endif
        // Cooperative A staging: 64 rows x 32 K f32 -> bf16 fragments in LDS.
#pragma unroll
        for (int e = 0; e < 8; ++e) {
            int idx = tid * 8 + e;
            int rl  = idx >> 5;          // local row 0..63
            int kk  = idx & 31;          // k within tile
            int m   = mblock + rl;
            int b   = m >> 10;           // SRCLEN = 1024
            int t   = m & 1023;
            int tok = src[b * SRCLEN + t];
            float val = emb[(size_t)tok * EMBED + kt * 32 + kk];
            int l2, i2;
            a_slot(rl, kk, &l2, &i2);
            aSh[rl >> 4][l2][i2] = f2bf(val);
        }
#if HAVE_TDM
        if (wave == 0) __builtin_amdgcn_s_wait_tensorcnt(0);
        __syncthreads();
        v16u bU = *(const v16u*)(&bSh[wave][lane * 16]);
#else
        __syncthreads();
        const u16* bptr = WxP + ((size_t)(ntile * (EMBED / 32) + kt)) * 512 + lane * 16;
        v16u bU = *(const v16u*)bptr;
#endif
#pragma unroll
        for (int mt = 0; mt < 4; ++mt) {
            v16u aU = *(const v16u*)(&aSh[mt][lane][0]);
            acc[mt] = wmma_bf16(aU, bU, acc[mt]);
        }
        __syncthreads();
    }

    // Epilogue: add bias, emit bf16 x_proj in [t][b][g] order.
    const int col = ntile * 16 + (lane & 15);
    const float bxv = bx[col];
#pragma unroll
    for (int mt = 0; mt < 4; ++mt) {
#pragma unroll
        for (int r = 0; r < 8; ++r) {
            int m = mblock + mt * 16 + ((lane < 16) ? r : (r + 8));
            int b = m >> 10;
            int t = m & 1023;
            xp[(size_t)t * BATCH * GATES + (size_t)b * GATES + col] =
                f2bf(acc[mt][r] + bxv);
        }
    }
}

// ---------------------------------------------------------------------------
// One LSTM timestep. Grid: 64 WGs (16 hidden units each), block 256 (8 waves).
// Each wave runs TWO independent WMMA accumulator chains (gates g and g+2,
// same mtile) sharing one A-fragment stream -> 2x ILP on the recurrence
// critical path, half the A traffic.
// ---------------------------------------------------------------------------
__global__ void __launch_bounds__(256)
lstm_step_kernel(int t, const u16* __restrict__ xp, const u16* __restrict__ WhP,
                 const u16* __restrict__ hA_rd, u16* __restrict__ hA_wr,
                 float* __restrict__ hbuf, float* __restrict__ cbuf) {
    __shared__ float gates_s[4][64][16];   // 16 KB

    const int tid  = threadIdx.x;
    const int lane = tid & 31;
    const int wave = tid >> 5;
    const int wg   = blockIdx.x;           // hidden slice [wg*16, wg*16+16)

    const int mt    = wave & 3;            // M-tile 0..3
    const int gate0 = wave >> 2;           // gates {0,2} or {1,3}
    const int gate1 = gate0 + 2;

    const u16* abase = hA_rd + ((size_t)(mt * (HIDDEN / 32))) * 512 + lane * 16;
    const u16* b0    = WhP + ((size_t)(gate0 * (HIDDEN / 16) + wg) * (HIDDEN / 32)) * 512 + lane * 16;
    const u16* b1    = WhP + ((size_t)(gate1 * (HIDDEN / 16) + wg) * (HIDDEN / 32)) * 512 + lane * 16;

    v8f acc0 = vzero8();
    v8f acc1 = vzero8();
#pragma unroll 4
    for (int k = 0; k < HIDDEN / 32; ++k) {
        v16u aU  = *(const v16u*)(abase + (size_t)k * 512);
        v16u bU0 = *(const v16u*)(b0 + (size_t)k * 512);
        v16u bU1 = *(const v16u*)(b1 + (size_t)k * 512);
        int kn = (k + 1 < HIDDEN / 32) ? (k + 1) : k;
        __builtin_prefetch(b0 + (size_t)kn * 512, 0, 0);
        __builtin_prefetch(b1 + (size_t)kn * 512, 0, 0);
        acc0 = wmma_bf16(aU, bU0, acc0);
        acc1 = wmma_bf16(aU, bU1, acc1);
    }
#pragma unroll
    for (int r = 0; r < 8; ++r) {
        int row = mt * 16 + ((lane < 16) ? r : (r + 8));
        gates_s[gate0][row][lane & 15] = acc0[r];
        gates_s[gate1][row][lane & 15] = acc1[r];
    }
    __syncthreads();

    // Elementwise gate math: 1024 (b, j) cells, 4 per thread.
    const u16* xpt = xp + (size_t)t * BATCH * GATES;
#pragma unroll
    for (int q = 0; q < 4; ++q) {
        int idx = tid + q * 256;
        int b   = idx >> 4;
        int jj  = idx & 15;
        int j   = wg * 16 + jj;

        float iv = sigmoidf(gates_s[0][b][jj] + bf2f(xpt[b * GATES + 0 * HIDDEN + j]));
        float fv = sigmoidf(gates_s[1][b][jj] + bf2f(xpt[b * GATES + 1 * HIDDEN + j]));
        float ov = sigmoidf(gates_s[2][b][jj] + bf2f(xpt[b * GATES + 2 * HIDDEN + j]));
        float gv = tanhf   (gates_s[3][b][jj] + bf2f(xpt[b * GATES + 3 * HIDDEN + j]));

        float c_old = cbuf[b * HIDDEN + j];
        float c_new = fv * c_old + iv * gv;
        float h_new = ov * tanhf(c_new);

        cbuf[b * HIDDEN + j] = c_new;
        hbuf[b * HIDDEN + j] = h_new;

        // Repack h_new into bf16 A-fragment layout for next step's WMMA.
        int mtile = b >> 4;
        int ktile = j >> 5;
        int kk    = j & 31;
        int l2, i2;
        a_slot(b, kk, &l2, &i2);
        hA_wr[((size_t)(mtile * (HIDDEN / 32) + ktile)) * 512 + l2 * 16 + i2] = f2bf(h_new);
    }
}

// ---------------------------------------------------------------------------
// Launch
// ---------------------------------------------------------------------------
extern "C" void kernel_launch(void* const* d_in, const int* in_sizes, int n_in,
                              void* d_out, int out_size, void* d_ws, size_t ws_size,
                              hipStream_t stream) {
    const int*   src = (const int*)  d_in[0];
    const float* emb = (const float*)d_in[1];
    const float* Wx  = (const float*)d_in[2];
    const float* bx  = (const float*)d_in[3];
    const float* Wh  = (const float*)d_in[4];

    float* out  = (float*)d_out;
    float* hbuf = out;                    // (B, H) final h
    float* cbuf = out + BATCH * HIDDEN;   // (B, H) final c

    // Workspace layout (bytes):
    //   [0,   4MB)   WxP : 256 ntiles x 16 ktiles x 1KB fragments
    //   [4MB, 12MB)  WhP : 256 ntiles x 32 ktiles x 1KB fragments
    //   [12MB,+256KB) hA : double-buffered packed bf16 h (2 x 128 KB)
    //   [13MB,+512MB) xp : bf16 x_proj [T][B][4H]
    char* ws  = (char*)d_ws;
    u16*  WxP = (u16*)(ws);
    u16*  WhP = (u16*)(ws + (size_t)4  * 1024 * 1024);
    u16*  hA  = (u16*)(ws + (size_t)12 * 1024 * 1024);
    u16*  xp  = (u16*)(ws + (size_t)13 * 1024 * 1024);

    // 1) Pack weights to bf16 fragment blobs.
    {
        int totWx = (GATES / 16) * (EMBED  / 32) * 32;   // 131072
        int totWh = (GATES / 16) * (HIDDEN / 32) * 32;   // 262144
        pack_w_kernel<<<(totWx + 255) / 256, 256, 0, stream>>>(Wx, WxP, GATES / 16, EMBED  / 32, EMBED);
        pack_w_kernel<<<(totWh + 255) / 256, 256, 0, stream>>>(Wh, WhP, GATES / 16, HIDDEN / 32, HIDDEN);
    }

    // 2) Zero h/c state and both packed-h buffers.
    init_state_kernel<<<(2 * BATCH * HIDDEN + 255) / 256, 256, 0, stream>>>(hA, out);

    // 3) Input projection for all timesteps (big parallel WMMA GEMM + TDM).
    {
        dim3 grid(BT / 64, GATES / 128);
        xproj_kernel<<<grid, 256, 0, stream>>>(src, emb, WxP, bx, xp);
    }

    // 4) Sequential recurrence: one launch per step (kernel boundary = sync).
    for (int t = 0; t < SRCLEN; ++t) {
        const u16* hA_rd = hA + (size_t)(t & 1)       * BATCH * HIDDEN;
        u16*       hA_wr = hA + (size_t)((t + 1) & 1) * BATCH * HIDDEN;
        lstm_step_kernel<<<HIDDEN / 16, 256, 0, stream>>>(t, xp, WhP, hA_rd, hA_wr, hbuf, cbuf);
    }
}
